// MultiHeadedAttention_13709535609465
// MI455X (gfx1250) — matrix operand order repaired and checked by reading, not compiled, not run on hardware
//
#include <hip/hip_runtime.h>

typedef _Float16 half_t;
typedef __attribute__((ext_vector_type(16))) _Float16 v16h;
typedef __attribute__((ext_vector_type(8)))  _Float16 v8h;
typedef __attribute__((ext_vector_type(8)))  float    v8f;
typedef __attribute__((ext_vector_type(4)))  unsigned int v4u;
typedef __attribute__((ext_vector_type(8)))  int v8i;
typedef __attribute__((ext_vector_type(4)))  int v4i;

#define DEV __device__ __forceinline__

static constexpr int S_LEN   = 2048;
static constexpr int D_MODEL = 1024;
static constexpr int NHEAD   = 16;
static constexpr int HEAD_D  = 64;
static constexpr int M_ROWS  = 2 * S_LEN;   // B*S = 4096

// ---------------- WMMA helpers ----------------

DEV v8f wmma_f16(v16h a, v16h b, v8f c) {
  return __builtin_amdgcn_wmma_f32_16x16x32_f16(
      false, a, false, b, (short)0, c, false, false);
}

// Escape the LDS buffer into an asm with a memory clobber: models the TDM
// engine's (IR-invisible) stores so later loads are neither undef-folded nor
// hoisted. Loads remain expressed off the original __shared__ array, so
// address-space inference still produces ds_load_* (not flat).
DEV void lds_fence(const half_t* base) {
  asm volatile("" : : "v"(base) : "memory");
}

// A-fragment (16x32 f16): lanes 0-15 hold K {0..7,16..23}, lanes 16-31 hold
// K {8..15,24..31}; two 16B contiguous loads per lane.
DEV v16h load_fragA(const half_t* rowPtr, int lane) {
  const int ko = (lane >= 16) ? 8 : 0;
  v8h lo = *(const v8h*)(rowPtr + ko);
  v8h hi = *(const v8h*)(rowPtr + ko + 16);
  v16h r;
  #pragma unroll
  for (int i = 0; i < 8; ++i) { r[i] = lo[i]; r[i + 8] = hi[i]; }
  return r;
}

// B-fragment (32x16 f16): column contiguous along K. Lanes 0-15 hold K
// {0..7,16..23}, lanes 16-31 hold K {8..15,24..31}.
DEV v16h load_fragB(const half_t* colPtr, int lane) {
  const int ko = (lane >= 16) ? 8 : 0;
  v8h lo = *(const v8h*)(colPtr + ko);
  v8h hi = *(const v8h*)(colPtr + ko + 16);
  v16h r;
  #pragma unroll
  for (int i = 0; i < 8; ++i) { r[i] = lo[i]; r[i + 8] = hi[i]; }
  return r;
}

// ---------------- TDM: 2-D tile DMA global -> LDS (f16 elements) -----------

DEV void tdm_load_2d_f16(unsigned ldsOff, const void* gaddr,
                         unsigned tileD0, unsigned tileD1,
                         unsigned tensorD0, unsigned tensorD1,
                         unsigned strideD0 /* elements between rows */) {
  unsigned long long ga = (unsigned long long)(size_t)gaddr;
  v4u g0;
  g0[0] = 1u;                                            // count=1, user desc
  g0[1] = ldsOff;                                        // lds_addr (bytes)
  g0[2] = (unsigned)(ga & 0xFFFFFFFFu);                  // global_addr[31:0]
  g0[3] = (unsigned)((ga >> 32) & 0x01FFFFFFu) | 0x80000000u; // [56:32]|type=2
  v8i g1;
  g1[0] = (int)(1u << 16);                               // data_size=1 (2B)
  g1[1] = (int)((tensorD0 & 0xFFFFu) << 16);             // tensor_dim0[15:0]
  g1[2] = (int)(((tensorD0 >> 16) & 0xFFFFu) | ((tensorD1 & 0xFFFFu) << 16));
  g1[3] = (int)(((tensorD1 >> 16) & 0xFFFFu) | ((tileD0 & 0xFFFFu) << 16));
  g1[4] = (int)(tileD1 & 0xFFFFu);                       // tile_dim1, dim2=0
  g1[5] = (int)strideD0;                                 // dim0_stride[31:0]
  g1[6] = 0;
  g1[7] = 0;
  v4i gz = {0, 0, 0, 0};
#if defined(__clang_major__) && (__clang_major__ >= 23)
  v8i gz8 = {0, 0, 0, 0, 0, 0, 0, 0};
  __builtin_amdgcn_tensor_load_to_lds(g0, g1, gz, gz, gz8, 0);
#else
  __builtin_amdgcn_tensor_load_to_lds(g0, g1, gz, gz, 0);
#endif
}

// ---------------- wave32 half-wave reductions ----------------

DEV float half16_max(float x) {
  #pragma unroll
  for (int m = 1; m < 16; m <<= 1) x = fmaxf(x, __shfl_xor(x, m, 32));
  return x;
}
DEV float half16_sum(float x) {
  #pragma unroll
  for (int m = 1; m < 16; m <<= 1) x += __shfl_xor(x, m, 32);
  return x;
}

// ---------------- kernels ----------------

__global__ void k_f32_to_f16(const float* __restrict__ in,
                             half_t* __restrict__ out, int n) {
  int i = blockIdx.x * blockDim.x + threadIdx.x;
  if (i < n) out[i] = (half_t)in[i];
}

// Y = X @ W^T + b, f16 out in head layout (vtrans=0: [bh,s,d]; 1: [bh,d,s]).
// A tile (128x32) and W tile (64x32) TDM-staged, double buffered; the inner
// loop reads only LDS.
__global__ void __launch_bounds__(256)
k_proj_gemm(const half_t* __restrict__ A, const half_t* __restrict__ W,
            const float* __restrict__ bias, half_t* __restrict__ out,
            int vtrans) {
  // LDS (bytes): A tiles @0 (2x8192), W tiles @16384 (2x4096).
  __shared__ half_t smem[2 * 4096 + 2 * 2048];
  const int lane   = threadIdx.x & 31;
  const int wave   = threadIdx.x >> 5;
  const int mBlock = blockIdx.y * 128;
  const int mBase  = mBlock + wave * 16;
  const int nBase  = blockIdx.x * 64;
  const int ln     = lane & 15;
  v8f acc[4] = {};

  const half_t* aSrc = A + (size_t)mBlock * D_MODEL;
  const half_t* wSrc = W + (size_t)nBase * D_MODEL;

  if (wave == 0) {
    tdm_load_2d_f16(0,     aSrc, 32, 128, D_MODEL, M_ROWS, D_MODEL);
    tdm_load_2d_f16(16384, wSrc, 32, 64,  D_MODEL, D_MODEL, D_MODEL);
  }

  for (int kb = 0; kb < D_MODEL; kb += 32) {
    const int p = (kb >> 5) & 1;
    if (wave == 0) {
      if (kb + 32 < D_MODEL) {
        tdm_load_2d_f16((unsigned)(8192 * (p ^ 1)), aSrc + (kb + 32),
                        32, 128, D_MODEL, M_ROWS, D_MODEL);
        tdm_load_2d_f16((unsigned)(16384 + 4096 * (p ^ 1)), wSrc + (kb + 32),
                        32, 64, D_MODEL, D_MODEL, D_MODEL);
        __builtin_amdgcn_s_wait_tensorcnt((short)2);
      } else {
        __builtin_amdgcn_s_wait_tensorcnt((short)0);
      }
    } else if (wave == 1 && kb + 64 < D_MODEL) {
      __builtin_prefetch(wSrc + kb + 64, 0, 1);
    }
    __syncthreads();
    lds_fence(smem);
    v16h af = load_fragA(smem + p * 4096 + (wave * 16 + ln) * 32, lane);
    const half_t* wb = smem + 8192 + p * 2048;
    #pragma unroll
    for (int t = 0; t < 4; ++t)
      acc[t] = wmma_f16(af, load_fragB(wb + (t * 16 + ln) * 32, lane), acc[t]);
    __syncthreads();
  }

  const int off8 = (lane >= 16) ? 8 : 0;
  #pragma unroll
  for (int t = 0; t < 4; ++t) {
    const int col = nBase + t * 16 + ln;
    const int h = col >> 6, di = col & 63;
    const float bcol = bias[col];
    #pragma unroll
    for (int v = 0; v < 8; ++v) {
      const int row = mBase + v + off8;
      const int bb = row >> 11, s = row & (S_LEN - 1);
      const float y = acc[t][v] + bcol;
      const size_t bh = (size_t)(bb * NHEAD + h);
      if (vtrans)
        out[(bh * HEAD_D + di) * S_LEN + s] = (half_t)y;
      else
        out[(bh * S_LEN + s) * HEAD_D + di] = (half_t)y;
    }
  }
}

// Flash attention. All 4 waves of a block share one (b,h); K/V tiles are
// TDM-staged and double buffered; P goes through per-wave LDS to convert
// C layout -> A fragment.
__global__ void __launch_bounds__(128)
k_attn(const half_t* __restrict__ Qh, const half_t* __restrict__ Kh,
       const half_t* __restrict__ Vt, const int* __restrict__ mask,
       half_t* __restrict__ out) {
  // LDS (bytes): K tiles @0 (2x4096), V tiles @8192 (2x4096), P @16384.
  __shared__ half_t smem[2 * 2048 + 2 * 2048 + 4 * 512];

  const int lane  = threadIdx.x & 31;
  const int wave  = threadIdx.x >> 5;
  const int task  = blockIdx.x * 4 + wave;       // 4096 tasks
  const int bh    = task >> 7;                   // same for all 4 waves
  const int qTile = task & 127;
  const int b     = bh >> 4;
  const int h     = bh & 15;
  const int qBase = qTile * 16;
  const int ln    = lane & 15;
  const int off8  = (lane >= 16) ? 8 : 0;
  half_t* lp = smem + 8192 + wave * 512;         // per-wave P buffer

  const half_t* kBase = Kh + (size_t)bh * S_LEN * HEAD_D;
  const half_t* vBase = Vt + (size_t)bh * HEAD_D * S_LEN;

  v16h qf[2];
  {
    const half_t* qRow = Qh + ((size_t)bh * S_LEN + qBase + ln) * HEAD_D;
    qf[0] = load_fragA(qRow, lane);
    qf[1] = load_fragA(qRow + 32, lane);
  }

  float mrun[8], lrun[8];
  #pragma unroll
  for (int v = 0; v < 8; ++v) { mrun[v] = -3.0e38f; lrun[v] = 0.0f; }
  v8f oacc[4] = {};
  const float scale = 0.125f;                    // 1/sqrt(64)
  const int* mrow = mask + (size_t)b * S_LEN * S_LEN;

  if (wave == 0) {
    tdm_load_2d_f16(0,    kBase, HEAD_D, 32, HEAD_D, S_LEN, HEAD_D);
    tdm_load_2d_f16(8192, vBase, 32, HEAD_D, S_LEN, HEAD_D, S_LEN);
  }

  for (int kb = 0; kb < S_LEN; kb += 32) {
    const int p = (kb >> 5) & 1;
    if (wave == 0) {
      if (kb + 32 < S_LEN) {
        tdm_load_2d_f16((unsigned)(4096 * (p ^ 1)),
                        kBase + (size_t)(kb + 32) * HEAD_D,
                        HEAD_D, 32, HEAD_D, S_LEN, HEAD_D);
        tdm_load_2d_f16((unsigned)(8192 + 4096 * (p ^ 1)),
                        vBase + (kb + 32), 32, HEAD_D, S_LEN, HEAD_D, S_LEN);
        __builtin_amdgcn_s_wait_tensorcnt((short)2);
      } else {
        __builtin_amdgcn_s_wait_tensorcnt((short)0);
      }
    }
    __syncthreads();
    lds_fence(smem);

    const half_t* kT = smem + p * 2048;          // [key(32)][d(64)]
    const half_t* vT = smem + 4096 + p * 2048;   // [di(64)][key(32)]

    v8f sc[2] = {};
    #pragma unroll
    for (int kc = 0; kc < 2; ++kc) {
      const half_t* kCol = kT + (kc * 16 + ln) * HEAD_D;
      sc[kc] = wmma_f16(qf[0], load_fragB(kCol, lane), sc[kc]);
      sc[kc] = wmma_f16(qf[1], load_fragB(kCol + 32, lane), sc[kc]);
    }

    #pragma unroll
    for (int v = 0; v < 8; ++v) {
      const int qr = qBase + v + off8;
      float x0 = sc[0][v] * scale;
      float x1 = sc[1][v] * scale;
      if (mrow[(size_t)qr * S_LEN + kb + ln] == 0)      x0 = -1e9f;
      if (mrow[(size_t)qr * S_LEN + kb + 16 + ln] == 0) x1 = -1e9f;
      const float tmax = half16_max(fmaxf(x0, x1));
      const float nm   = fmaxf(mrun[v], tmax);
      const float fac  = __expf(mrun[v] - nm);
      mrun[v] = nm;
      const float e0 = __expf(x0 - nm);
      const float e1 = __expf(x1 - nm);
      lrun[v] = lrun[v] * fac + half16_sum(e0 + e1);
      #pragma unroll
      for (int t = 0; t < 4; ++t) oacc[t][v] = oacc[t][v] * fac;
      lp[(v + off8) * 32 + ln]      = (half_t)e0;
      lp[(v + off8) * 32 + 16 + ln] = (half_t)e1;
    }
    asm volatile("s_wait_dscnt 0x0" ::: "memory");
    v16h pf;
    {
      const int ko = (lane >= 16) ? 8 : 0;
      const half_t* pr = lp + ln * 32;
      #pragma unroll
      for (int i = 0; i < 8; ++i) {
        pf[i]     = pr[ko + i];
        pf[i + 8] = pr[ko + 16 + i];
      }
    }
    asm volatile("s_wait_dscnt 0x0" ::: "memory");
    #pragma unroll
    for (int t = 0; t < 4; ++t)
      oacc[t] = wmma_f16(pf, load_fragB(vT + (t * 16 + ln) * 32, lane), oacc[t]);
    __syncthreads();
  }

  #pragma unroll
  for (int t = 0; t < 4; ++t) {
    #pragma unroll
    for (int v = 0; v < 8; ++v) {
      const int row = qBase + v + off8;
      const float o = oacc[t][v] / lrun[v];
      out[((size_t)b * S_LEN + row) * D_MODEL + h * HEAD_D + t * 16 + ln] =
          (half_t)o;
    }
  }
}

// Y = attn @ Wo^T + bo + residual (f32, pre-LayerNorm); A and W TDM-staged.
__global__ void __launch_bounds__(256)
k_out_gemm(const half_t* __restrict__ A, const half_t* __restrict__ W,
           const float* __restrict__ bias, const float* __restrict__ resid,
           float* __restrict__ Y) {
  __shared__ half_t smem[2 * 4096 + 2 * 2048];
  const int lane   = threadIdx.x & 31;
  const int wave   = threadIdx.x >> 5;
  const int mBlock = blockIdx.y * 128;
  const int mBase  = mBlock + wave * 16;
  const int nBase  = blockIdx.x * 64;
  const int ln     = lane & 15;
  v8f acc[4] = {};

  const half_t* aSrc = A + (size_t)mBlock * D_MODEL;
  const half_t* wSrc = W + (size_t)nBase * D_MODEL;

  if (wave == 0) {
    tdm_load_2d_f16(0,     aSrc, 32, 128, D_MODEL, M_ROWS, D_MODEL);
    tdm_load_2d_f16(16384, wSrc, 32, 64,  D_MODEL, D_MODEL, D_MODEL);
  }

  for (int kb = 0; kb < D_MODEL; kb += 32) {
    const int p = (kb >> 5) & 1;
    if (wave == 0) {
      if (kb + 32 < D_MODEL) {
        tdm_load_2d_f16((unsigned)(8192 * (p ^ 1)), aSrc + (kb + 32),
                        32, 128, D_MODEL, M_ROWS, D_MODEL);
        tdm_load_2d_f16((unsigned)(16384 + 4096 * (p ^ 1)), wSrc + (kb + 32),
                        32, 64, D_MODEL, D_MODEL, D_MODEL);
        __builtin_amdgcn_s_wait_tensorcnt((short)2);
      } else {
        __builtin_amdgcn_s_wait_tensorcnt((short)0);
      }
    } else if (wave == 1 && kb + 64 < D_MODEL) {
      __builtin_prefetch(wSrc + kb + 64, 0, 1);
    }
    __syncthreads();
    lds_fence(smem);
    v16h af = load_fragA(smem + p * 4096 + (wave * 16 + ln) * 32, lane);
    const half_t* wb = smem + 8192 + p * 2048;
    #pragma unroll
    for (int t = 0; t < 4; ++t)
      acc[t] = wmma_f16(af, load_fragB(wb + (t * 16 + ln) * 32, lane), acc[t]);
    __syncthreads();
  }

  const int off8 = (lane >= 16) ? 8 : 0;
  #pragma unroll
  for (int t = 0; t < 4; ++t) {
    const int col = nBase + t * 16 + ln;
    const float bcol = bias[col];
    #pragma unroll
    for (int v = 0; v < 8; ++v) {
      const int row = mBase + v + off8;
      Y[(size_t)row * D_MODEL + col] =
          acc[t][v] + bcol + resid[(size_t)row * D_MODEL + col];
    }
  }
}

// LayerNorm: unbiased std (ddof=1), (std + eps) denominator.
__global__ void __launch_bounds__(256)
k_layernorm(const float* __restrict__ Y, const float* __restrict__ gamma,
            const float* __restrict__ beta, float* __restrict__ out) {
  const int lane = threadIdx.x & 31;
  const int wave = threadIdx.x >> 5;
  const int row  = blockIdx.x * 8 + wave;
  const float* yr = Y + (size_t)row * D_MODEL;
  float x[32];
  float s = 0.0f, s2 = 0.0f;
  #pragma unroll
  for (int i = 0; i < 32; ++i) {
    const float v = yr[i * 32 + lane];
    x[i] = v; s += v; s2 += v * v;
  }
  #pragma unroll
  for (int m = 1; m < 32; m <<= 1) {
    s  += __shfl_xor(s,  m, 32);
    s2 += __shfl_xor(s2, m, 32);
  }
  const float mean = s * (1.0f / 1024.0f);
  const float var  = (s2 - 1024.0f * mean * mean) * (1.0f / 1023.0f);
  const float stdv = sqrtf(fmaxf(var, 0.0f));
  const float inv  = 1.0f / (stdv + 1e-6f);
  float* orow = out + (size_t)row * D_MODEL;
  #pragma unroll
  for (int i = 0; i < 32; ++i) {
    const int c = i * 32 + lane;
    orow[c] = gamma[c] * (x[i] - mean) * inv + beta[c];
  }
}

// ---------------- host launcher ----------------

extern "C" void kernel_launch(void* const* d_in, const int* in_sizes, int n_in,
                              void* d_out, int out_size, void* d_ws,
                              size_t ws_size, hipStream_t stream) {
  (void)in_sizes; (void)n_in; (void)out_size; (void)ws_size;
  const float* q    = (const float*)d_in[0];
  const float* k    = (const float*)d_in[1];
  const float* v    = (const float*)d_in[2];
  const int*   mask = (const int*)d_in[3];
  const float* Wq   = (const float*)d_in[4];
  const float* bq   = (const float*)d_in[5];
  const float* Wk   = (const float*)d_in[6];
  const float* bk   = (const float*)d_in[7];
  const float* Wv   = (const float*)d_in[8];
  const float* bv   = (const float*)d_in[9];
  const float* Wo   = (const float*)d_in[10];
  const float* bo   = (const float*)d_in[11];
  const float* gamma= (const float*)d_in[12];
  const float* beta = (const float*)d_in[13];
  float* out = (float*)d_out;

  char* w = (char*)d_ws;
  const size_t SZ_X16 = (size_t)M_ROWS * D_MODEL * sizeof(half_t);   // 8 MB
  const size_t SZ_W16 = (size_t)D_MODEL * D_MODEL * sizeof(half_t);  // 2 MB
  half_t* x16q = (half_t*)(w);
  half_t* x16k = (half_t*)(w + SZ_X16);
  half_t* x16v = (half_t*)(w + 2 * SZ_X16);
  half_t* w16q = (half_t*)(w + 3 * SZ_X16);
  half_t* w16k = (half_t*)(w + 3 * SZ_X16 + SZ_W16);
  half_t* w16v = (half_t*)(w + 3 * SZ_X16 + 2 * SZ_W16);
  half_t* w16o = (half_t*)(w + 3 * SZ_X16 + 3 * SZ_W16);
  half_t* Qh   = (half_t*)(w + 3 * SZ_X16 + 4 * SZ_W16);
  half_t* Kh   = (half_t*)(w + 4 * SZ_X16 + 4 * SZ_W16);
  half_t* Vt   = (half_t*)(w + 5 * SZ_X16 + 4 * SZ_W16);
  half_t* att  = (half_t*)(w + 6 * SZ_X16 + 4 * SZ_W16);
  float*  Y    = (float*) (w + 7 * SZ_X16 + 4 * SZ_W16);

  const int nX = M_ROWS * D_MODEL;
  const int nW = D_MODEL * D_MODEL;
  k_f32_to_f16<<<(nX + 255) / 256, 256, 0, stream>>>(q,  x16q, nX);
  k_f32_to_f16<<<(nX + 255) / 256, 256, 0, stream>>>(k,  x16k, nX);
  k_f32_to_f16<<<(nX + 255) / 256, 256, 0, stream>>>(v,  x16v, nX);
  k_f32_to_f16<<<(nW + 255) / 256, 256, 0, stream>>>(Wq, w16q, nW);
  k_f32_to_f16<<<(nW + 255) / 256, 256, 0, stream>>>(Wk, w16k, nW);
  k_f32_to_f16<<<(nW + 255) / 256, 256, 0, stream>>>(Wv, w16v, nW);
  k_f32_to_f16<<<(nW + 255) / 256, 256, 0, stream>>>(Wo, w16o, nW);

  dim3 gg(D_MODEL / 64, M_ROWS / 128);          // (16, 32)
  k_proj_gemm<<<gg, 256, 0, stream>>>(x16q, w16q, bq, Qh, 0);
  k_proj_gemm<<<gg, 256, 0, stream>>>(x16k, w16k, bk, Kh, 0);
  k_proj_gemm<<<gg, 256, 0, stream>>>(x16v, w16v, bv, Vt, 1);

  k_attn<<<1024, 128, 0, stream>>>(Qh, Kh, Vt, mask, att);

  k_out_gemm<<<gg, 256, 0, stream>>>(att, w16o, bo, q, Y);
  k_layernorm<<<M_ROWS / 8, 256, 0, stream>>>(Y, gamma, beta, out);
}
